// Self_Attention_19559281066068
// MI455X (gfx1250) — compile-verified, matrix operand order checked by reading, not run on hardware
//
#include <hip/hip_runtime.h>
#include <hip/hip_bf16.h>

typedef __attribute__((ext_vector_type(16))) __bf16 v16bf;
typedef __attribute__((ext_vector_type(8)))  __bf16 v8bf;
typedef __attribute__((ext_vector_type(4)))  __bf16 v4bf;
typedef __attribute__((ext_vector_type(8)))  float  v8f;

#define N_NODES 262144
#define DIM 512
#define HID 512
#define NSEG 1024
#define LDS_STRIDE 520   // 512 + 8 bf16 pad -> row stride 1040B, spreads LDS banks

// Branch-free tanh: tanh(x) = 1 - 2/(exp(2x)+1).
// exp(2x)->0 gives -1, exp(2x)->inf gives +1 (rcp(inf)=0). ~5 VALU, no exec
// divergence (the libm tanhf expands to a branchy 2-path version that costs
// ~30 VALU + saveexec per element).
__device__ __forceinline__ float fast_tanh(float x) {
  float ex = __expf(2.0f * x);
  return 1.0f - 2.0f * __builtin_amdgcn_rcpf(ex + 1.0f);
}

// ---------------------------------------------------------------------------
// Kernel 0: convert W [H][D] f32 -> bf16, pre-swizzled into WMMA B-fragment
// layout (K-tile-major), and zero the per-segment exp sums.
// Tile (kt,nt) covers B[k][j] = W[nt*16+j][kt*32+k]; lane L holds column
// j=(L&15); lanes 0-15 hold K 0-7 & 16-23, lanes 16-31 hold K 8-15 & 24-31.
// ---------------------------------------------------------------------------
__global__ void __launch_bounds__(512) prep_kernel(const float* __restrict__ W,
                                                   __bf16* __restrict__ Wb,
                                                   float* __restrict__ gsum) {
  int t = blockIdx.x * blockDim.x + threadIdx.x;     // 0 .. H*D-1 (262144)
  if (t < NSEG) gsum[t] = 0.0f;
  int tile   = t >> 9;            // 512 elements per 32x16 tile
  int within = t & 511;
  int lane   = within >> 4;       // which lane's fragment
  int e      = within & 15;       // element within the 16-elem fragment
  int kt     = tile >> 5;         // 16 K-tiles of 32
  int nt     = tile & 31;         // 32 N-tiles of 16
  int half   = lane >> 4;
  int j = nt * 16 + (lane & 15);
  int k = kt * 32 + ((e < 8) ? (e + 8 * half) : (e + 8 + 8 * half));
  Wb[t] = (__bf16)W[j * DIM + k];
}

// ---------------------------------------------------------------------------
// Kernel 1: per-node attention logit.
//   beta[n] = sum_j V[j] * tanh( sum_k embed[n][k] * W[j][k] )
// One wave owns a 16-row node tile. Stage the 16x512 f32 tile once into LDS
// as bf16, then 32 N-tiles x 16 K-steps of v_wmma_f32_16x16x32_bf16.
// Epilogue: tanh + V-weighting per lane, cross-lane reduce over the 16-lane
// halves, exp, store exp_beta, atomic-add into gsum[batch_index].
// ---------------------------------------------------------------------------
__global__ void __launch_bounds__(64) attn_beta_kernel(
    const float* __restrict__ embed, const int* __restrict__ bidx,
    const __bf16* __restrict__ Wb, const float* __restrict__ Vatt,
    float* __restrict__ exp_beta, float* __restrict__ gsum) {
  __shared__ __bf16 a_lds_all[2 * 16 * LDS_STRIDE];

  const int tid  = threadIdx.x;
  const int wave = tid >> 5;
  const int lane = tid & 31;
  const int half = lane >> 4;
  const int l15  = lane & 15;
  const int m0   = blockIdx.x * 32 + wave * 16;      // first node row of tile
  __bf16* a_lds  = a_lds_all + wave * 16 * LDS_STRIDE;

  // Stage 16 rows x 512 cols f32 -> bf16 into LDS (b128 global, b64 ds store)
  for (int i = lane; i < 16 * 128; i += 32) {        // 128 float4 per row
    int row = i >> 7;
    int c4  = (i & 127) << 2;
    const float4 f = *(const float4*)(embed + (size_t)(m0 + row) * DIM + c4);
    v4bf p;
    p[0] = (__bf16)f.x; p[1] = (__bf16)f.y; p[2] = (__bf16)f.z; p[3] = (__bf16)f.w;
    *(v4bf*)(a_lds + row * LDS_STRIDE + c4) = p;
  }
  __syncthreads();

  float bsum[8] = {0.f, 0.f, 0.f, 0.f, 0.f, 0.f, 0.f, 0.f};

  for (int nt = 0; nt < 32; ++nt) {
    v8f c = {};
    #pragma unroll 4
    for (int kt = 0; kt < 16; ++kt) {
      // A fragment: row = l15; lanes 0-15 take K k0+0..7 / k0+16..23,
      // lanes 16-31 take K k0+8..15 / k0+24..31  (ISA 7.12.2 16-bit A layout)
      const __bf16* ap = a_lds + l15 * LDS_STRIDE + kt * 32 + 8 * half;
      v8bf alo = *(const v8bf*)(ap);          // ds_load_b128
      v8bf ahi = *(const v8bf*)(ap + 16);     // ds_load_b128
      v16bf a;
      #pragma unroll
      for (int i = 0; i < 8; ++i) { a[i] = alo[i]; a[i + 8] = ahi[i]; }

      // B fragment: pre-swizzled, lane's 16 bf16 are contiguous (32B)
      v16bf b = *(const v16bf*)(Wb + ((((kt << 5) + nt) << 9) + (lane << 4)));

      c = __builtin_amdgcn_wmma_f32_16x16x32_bf16(
              false, a, false, b, (short)0, c, false, false);
    }
    // C layout: lane holds N=l15, M = r + 8*half for VGPR r
    float vj = Vatt[nt * 16 + l15];
    #pragma unroll
    for (int r = 0; r < 8; ++r) bsum[r] += fast_tanh(c[r]) * vj;
  }

  // Reduce over the 16 lanes of each half (sum over output columns j)
  #pragma unroll
  for (int r = 0; r < 8; ++r) {
    float s = bsum[r];
    s += __shfl_xor(s, 1, 32);
    s += __shfl_xor(s, 2, 32);
    s += __shfl_xor(s, 4, 32);
    s += __shfl_xor(s, 8, 32);
    bsum[r] = s;
  }

  if (l15 == 0) {   // lane 0 -> rows m0..m0+7, lane 16 -> rows m0+8..m0+15
    #pragma unroll
    for (int r = 0; r < 8; ++r) {
      int   row = m0 + half * 8 + r;
      float eb  = __expf(bsum[r]);
      exp_beta[row] = eb;
      __hip_atomic_fetch_add(&gsum[bidx[row]], eb,
                             __ATOMIC_RELAXED, __HIP_MEMORY_SCOPE_AGENT);
    }
  }
}

// ---------------------------------------------------------------------------
// Kernel 2: segment-weighted pooling. batch_index is sorted, so one
// workgroup per segment, bounds via binary search; alphas staged in LDS;
// 256 threads x float2 = one fully coalesced 2KB row read per node.
// ---------------------------------------------------------------------------
__device__ __forceinline__ int lower_bound_i(const int* __restrict__ a,
                                             int n, int key) {
  int lo = 0, hi = n;
  while (lo < hi) {
    int mid = (lo + hi) >> 1;
    if (a[mid] < key) lo = mid + 1; else hi = mid;
  }
  return lo;
}

__global__ void __launch_bounds__(256) pool_kernel(
    const float* __restrict__ embed, const int* __restrict__ bidx,
    const float* __restrict__ exp_beta, const float* __restrict__ gsum,
    float* __restrict__ out) {
  __shared__ int   seg[2];
  __shared__ float alpha_s[256];

  const int g   = blockIdx.x;
  const int tid = threadIdx.x;
  if (tid < 2) seg[tid] = lower_bound_i(bidx, N_NODES, g + tid);
  __syncthreads();
  const int start = seg[0];
  const int end   = seg[1];
  const float inv = 1.0f / gsum[g];        // unused when segment is empty

  float acc0 = 0.f, acc1 = 0.f;
  const int d = tid << 1;

  for (int c0 = start; c0 < end; c0 += 256) {
    int nn = end - c0; if (nn > 256) nn = 256;
    __syncthreads();
    if (tid < nn) alpha_s[tid] = exp_beta[c0 + tid] * inv;
    __syncthreads();
    for (int j = 0; j < nn; ++j) {
      float a = alpha_s[j];
      const float2 e2 = *(const float2*)(embed + (size_t)(c0 + j) * DIM + d);
      acc0 = fmaf(a, e2.x, acc0);
      acc1 = fmaf(a, e2.y, acc1);
    }
  }
  out[(size_t)g * DIM + d]     = acc0;
  out[(size_t)g * DIM + d + 1] = acc1;
}

// ---------------------------------------------------------------------------
extern "C" void kernel_launch(void* const* d_in, const int* in_sizes, int n_in,
                              void* d_out, int out_size, void* d_ws, size_t ws_size,
                              hipStream_t stream) {
  const float* embed = (const float*)d_in[0];   // [N, D] f32
  const int*   bidx  = (const int*)d_in[1];     // [N] sorted segment ids
  const float* W     = (const float*)d_in[2];   // [H, D] f32
  const float* Vatt  = (const float*)d_in[3];   // [1, H] f32
  float*       out   = (float*)d_out;           // [G, D] f32

  char* ws = (char*)d_ws;
  __bf16* Wb       = (__bf16*)ws;                                   // 512 KB
  float*  exp_beta = (float*)(ws + (512u << 10));                   // 1 MB
  float*  gsum     = (float*)(ws + (512u << 10) + (1024u << 10));   // 4 KB

  // Phase 0: W -> bf16 WMMA fragments, zero gsum
  prep_kernel<<<HID * DIM / 512, 512, 0, stream>>>(W, Wb, gsum);
  // Phase 1: exp(beta) per node + per-segment sums (WMMA GEMM, fused tanh/V)
  attn_beta_kernel<<<N_NODES / 32, 64, 0, stream>>>(embed, bidx, Wb, Vatt,
                                                    exp_beta, gsum);
  // Phase 2: alpha-weighted segment pooling
  pool_kernel<<<NSEG, 256, 0, stream>>>(embed, bidx, exp_beta, gsum, out);
}
